// SwinTransformerBlock_46239617908838
// MI455X (gfx1250) — matrix operand order repaired and checked by reading, not compiled, not run on hardware
//
#include <hip/hip_runtime.h>
#include <hip/hip_bf16.h>
#include <math.h>

// ---------------------------------------------------------------------------
// SwinTransformerBlock (CSWin/LePE style) for MI455X / gfx1250.
// All 1x1 convs + attention matmuls run on v_wmma_f32_16x16x32_f16 (wave32).
// Attention V tiles are staged LDS-side via the Tensor Data Mover when the
// toolchain exposes __builtin_amdgcn_tensor_load_to_lds (TENSORcnt-tracked).
// ---------------------------------------------------------------------------

#define NPIX  65536   // 256*256
#define HWDIM 256

typedef _Float16 half_t;
typedef __attribute__((ext_vector_type(16))) _Float16 v16h;
typedef __attribute__((ext_vector_type(8)))  _Float16 v8h;
typedef __attribute__((ext_vector_type(8)))  float    v8f;
typedef __attribute__((ext_vector_type(4)))  unsigned int u32x4;
typedef __attribute__((ext_vector_type(8)))  int i32x8;
typedef __attribute__((ext_vector_type(4)))  int i32x4;

union u16h { v16h v; struct { v8h lo, hi; } s; };

#if defined(__HIP_DEVICE_COMPILE__) && defined(__gfx1250__) && \
    __has_builtin(__builtin_amdgcn_tensor_load_to_lds) &&      \
    __has_builtin(__builtin_amdgcn_s_wait_tensorcnt)
#define HAVE_TDM 1
#else
#define HAVE_TDM 0
#endif

__device__ __forceinline__ v8f wmma_f16(v16h a, v16h b, v8f c) {
  // D = A(16x32 f16) * B(32x16 f16) + C(16x16 f32)
  return __builtin_amdgcn_wmma_f32_16x16x32_f16(
      /*neg_a=*/false, a, /*neg_b=*/false, b,
      /*c_mod=*/(short)0, c, /*reuse_a=*/false, /*reuse_b=*/false);
}

__device__ __forceinline__ float gelu_f(float x) {
  return 0.5f * x * (1.f + erff(x * 0.70710678118654752f));
}
__device__ __forceinline__ float sigmoid_f(float x) {
  return 1.f / (1.f + __expf(-x));
}

#if HAVE_TDM
// Issue a 1-D TDM copy of `words8` 8-byte words from global `gsrc` into LDS at
// byte offset `lds_off`. D# per CDNA5 ISA ch.8: group0 {count=1, lds_addr,
// global_addr, type=2}; group1 {data_size=8B, tensor_dim0, tile_dim0, stride}.
__device__ __forceinline__ void tdm_load_1d(unsigned lds_off, const void* gsrc,
                                            unsigned words8) {
  unsigned long long ga = (unsigned long long)(size_t)gsrc;
  unsigned ga_lo = (unsigned)__builtin_amdgcn_readfirstlane((int)(unsigned)ga);
  unsigned ga_hi = (unsigned)__builtin_amdgcn_readfirstlane((int)(unsigned)(ga >> 32));
  unsigned lo    = (unsigned)__builtin_amdgcn_readfirstlane((int)lds_off);
  unsigned nw    = (unsigned)__builtin_amdgcn_readfirstlane((int)words8);
  u32x4 g0;
  g0[0] = 1u;                          // count=1 (valid user descriptor)
  g0[1] = lo;                          // lds_addr
  g0[2] = ga_lo;                       // global_addr[31:0]
  g0[3] = ga_hi | (2u << 30);          // global_addr[56:32] | type=2 ("image")
  i32x8 g1 = {};
  g1[0] = (int)(3u << 16);             // data_size=3 (8 bytes), mask/flags=0
  g1[1] = (int)((nw & 0xFFFFu) << 16); // tensor_dim0[15:0]  @ bits 63:48
  g1[2] = (int)(nw >> 16);             // tensor_dim0[31:16] @ bits 79:64
  g1[3] = (int)((nw & 0xFFFFu) << 16); // tile_dim0          @ bits 127:112
  g1[5] = (int)nw;                     // tensor_dim0_stride[31:0]
  i32x4 gz = {};
#if __clang_major__ >= 23
  i32x8 gz8 = {};
  __builtin_amdgcn_tensor_load_to_lds(g0, g1, gz, gz, gz8, 0);
#else
  __builtin_amdgcn_tensor_load_to_lds(g0, g1, gz, gz, 0);
#endif
}
#endif

// ---------------------------------------------------------------------------
// f32 -> f16 convert (weights & staging)
__global__ void __launch_bounds__(256)
k_f32_to_f16(const float* __restrict__ in, half_t* __restrict__ out, int n) {
  int i = blockIdx.x * 256 + threadIdx.x;
  if (i < n) out[i] = (half_t)in[i];
}

// ---------------------------------------------------------------------------
// Channel LayerNorm over C=64 per pixel; emit pixel-major f16 [P,64] for WMMA B.
__global__ void __launch_bounds__(256)
k_ln_pack(const float* __restrict__ x, const float* __restrict__ g,
          const float* __restrict__ b, half_t* __restrict__ out) {
  int p = blockIdx.x * 256 + threadIdx.x;
  if (p >= NPIX) return;
  float s = 0.f, ss = 0.f;
  for (int c = 0; c < 64; ++c) {
    float v = x[(size_t)c * NPIX + p];
    s += v; ss += v * v;
  }
  float mu  = s  * (1.f / 64.f);
  float var = ss * (1.f / 64.f) - mu * mu;
  float inv = rsqrtf(var + 1e-6f);
  for (int c = 0; c < 64; ++c) {
    float v = (x[(size_t)c * NPIX + p] - mu) * inv * g[c] + b[c];
    out[(size_t)p * 64 + c] = (half_t)v;
  }
}

// ---------------------------------------------------------------------------
// WMMA GEMM: Y[M,N] (f32, channel-major) = W[M,K](f16) * X[N,K](f16)^T + bias
// 8 waves per block, one 16x16 tile per wave, software-pipelined K loop.
// K multiple of 32. act: 0 none, 1 sigmoid, 2 tanh.
__global__ void __launch_bounds__(256)
k_gemm16(const half_t* __restrict__ W, const half_t* __restrict__ X,
         const float* __restrict__ bias, float* __restrict__ Y,
         int M, int N, int K, int act) {
  int tile = blockIdx.x * 8 + (threadIdx.x >> 5);
  int mt   = tile % (M >> 4);
  int nt   = tile / (M >> 4);
  int lane = threadIdx.x & 31;
  const half_t* Wt = W + (size_t)(mt * 16 + (lane & 15)) * K;
  const half_t* Xt = X + (size_t)(nt * 16 + (lane & 15)) * K;
  int ka0 = (lane >> 4) << 3;   // A fragment: chunks k0, k0+16
  int kb0 = (lane >> 4) << 3;   // B fragment: chunks k0, k0+16
  u16h ua, ub;
  ua.s.lo = *(const v8h*)(Wt + ka0);
  ua.s.hi = *(const v8h*)(Wt + ka0 + 16);
  ub.s.lo = *(const v8h*)(Xt + kb0);
  ub.s.hi = *(const v8h*)(Xt + kb0 + 16);
  v8f acc = {};
  for (int kc = 0; kc < K; kc += 32) {
    u16h na = ua, nb = ub;
    if (kc + 32 < K) {                 // preload next fragments (pipelined)
      int ka = kc + 32 + ka0, kb = kc + 32 + kb0;
      na.s.lo = *(const v8h*)(Wt + ka);
      na.s.hi = *(const v8h*)(Wt + ka + 16);
      nb.s.lo = *(const v8h*)(Xt + kb);
      nb.s.hi = *(const v8h*)(Xt + kb + 16);
      __builtin_prefetch(Xt + kc + 64, 0, 0);  // global_prefetch_b8
    }
    acc = wmma_f16(ua.v, ub.v, acc);
    ua = na; ub = nb;
  }
  int n     = nt * 16 + (lane & 15);
  int mbase = mt * 16 + ((lane >> 4) << 3);
  for (int r = 0; r < 8; ++r) {
    int m = mbase + r;
    float v = acc[r] + bias[m];
    if (act == 1)      v = sigmoid_f(v);
    else if (act == 2) v = tanhf(v);
    Y[(size_t)m * N + n] = v;
  }
}

// ---------------------------------------------------------------------------
// Depthwise KHxKW conv, SAME padding, optional pre-add input and GELU output.
__global__ void __launch_bounds__(256)
k_dwconv(const float* __restrict__ in, const float* __restrict__ add,
         const float* __restrict__ w, const float* __restrict__ b,
         float* __restrict__ out, int C, int KH, int KW, int act) {
  int tid = blockIdx.x * 256 + threadIdx.x;
  if (tid >= C * NPIX) return;
  int c = tid >> 16;
  int p = tid & (NPIX - 1);
  int h = p >> 8, x = p & 255;
  int ph = KH >> 1, pw = KW >> 1;
  float acc = b[c];
  const float* wc = w + c * KH * KW;
  for (int i = 0; i < KH; ++i) {
    int hh = h + i - ph;
    if ((unsigned)hh >= HWDIM) continue;
    for (int j = 0; j < KW; ++j) {
      int xx = x + j - pw;
      if ((unsigned)xx >= HWDIM) continue;
      int q = (c << 16) + (hh << 8) + xx;
      float v = in[q];
      if (add) v += add[q];
      acc += v * wc[i * KW + j];
    }
  }
  out[tid] = act ? gelu_f(acc) : acc;
}

// ---------------------------------------------------------------------------
// Pack channel-major f32 [C,P] into pixel-major f16 [P,Wtot] at column colOff,
// optional elementwise multiplier (for h*r in the gate).
__global__ void __launch_bounds__(256)
k_pack_pm(const float* __restrict__ in, const float* __restrict__ mul,
          half_t* __restrict__ out, int C, int Wtot, int colOff) {
  int tid = blockIdx.x * 256 + threadIdx.x;
  if (tid >= C * NPIX) return;
  int c = tid % C;
  int p = tid / C;
  float v = in[(size_t)c * NPIX + p];
  if (mul) v *= mul[(size_t)c * NPIX + p];
  out[(size_t)p * Wtot + colOff + c] = (half_t)v;
}

// ---------------------------------------------------------------------------
// Window packing with shift-roll folded into index math.
// qw/kw: [win][tok][32] f16 (q pre-scaled by C^-0.5), vw: [win][32][tok] f16.
__global__ void __launch_bounds__(256)
k_attn_pack(const float* __restrict__ q5, const float* __restrict__ k5,
            const float* __restrict__ lepe, const float* __restrict__ v,
            half_t* __restrict__ qw, half_t* __restrict__ kw,
            half_t* __restrict__ vw, int hsp, int wsp, float scale) {
  int t = blockIdx.x * 256 + threadIdx.x;   // over nwin*T == NPIX
  if (t >= NPIX) return;
  int T   = hsp * wsp;
  int win = t / T, tok = t % T;
  int wpr = HWDIM / wsp;
  int hr = (win / wpr) * hsp + tok / wsp;
  int wr = (win % wpr) * wsp + tok % wsp;
  int h  = (hr + (hsp >> 1)) & 255;
  int x  = (wr + (wsp >> 1)) & 255;
  int p  = (h << 8) | x;
  for (int c = 0; c < 32; ++c) {
    float lp = lepe[(size_t)c * NPIX + p];
    qw[(size_t)t * 32 + c] = (half_t)((q5[(size_t)c * NPIX + p] + lp) * scale);
    kw[(size_t)t * 32 + c] = (half_t)(k5[(size_t)c * NPIX + p] + lp);
    vw[((size_t)win * 32 + c) * T + tok] = (half_t)v[(size_t)c * NPIX + p];
  }
}

// ---------------------------------------------------------------------------
// Windowed attention, one wave per 16 query rows.
//   0) TDM: DMA the window's V tile ([32][T] f16) into LDS (overlaps phase 1)
//   1) S = qK^T via WMMA, scores spilled to LDS (f16)
//   2) in-LDS row softmax
//   3) s_wait_tensorcnt 0; O = P*V via WMMA (P from LDS in A-layout, V from LDS)
//   4) store with +lepe and inverse roll
__global__ void __launch_bounds__(32)
k_attn(const half_t* __restrict__ qw, const half_t* __restrict__ kw,
       const half_t* __restrict__ vw, const float* __restrict__ lepe,
       float* __restrict__ out, int hsp, int wsp) {
  __shared__ half_t sS[16 * 1024];          // 32 KB scores
#if HAVE_TDM
  __shared__ half_t vL[32 * 1024];          // 64 KB V tile (TDM destination)
#endif
  int T    = hsp * wsp;
  int tpw  = T >> 4;
  int win  = blockIdx.x / tpw;
  int qb   = (blockIdx.x % tpw) << 4;
  int lane = threadIdx.x & 31;
  const half_t* vbase = vw + (size_t)win * 32 * T;

#if HAVE_TDM
  // Kick the V-tile DMA first: 32*T halves = T*8 eight-byte words.
  tdm_load_1d((unsigned)(size_t)&vL[0], vbase, (unsigned)(T * 8));
#endif

  // q fragment (A-layout), loaded once
  const half_t* qrow = qw + ((size_t)win * T + qb + (lane & 15)) * 32;
  u16h ua;
  int ka = (lane >> 4) << 3;
  ua.s.lo = *(const v8h*)(qrow + ka);
  ua.s.hi = *(const v8h*)(qrow + ka + 16);

  // ---- S = q @ k^T, spilled to LDS as f16
  for (int kt = 0; kt < T; kt += 16) {
    const half_t* krow = kw + ((size_t)win * T + kt + (lane & 15)) * 32;
    u16h ub;
    int kb = (lane >> 4) << 3;
    ub.s.lo = *(const v8h*)(krow + kb);
    ub.s.hi = *(const v8h*)(krow + kb + 16);
    v8f sc = {};
    sc = wmma_f16(ua.v, ub.v, sc);
    int n  = kt + (lane & 15);
    int mb = (lane >> 4) << 3;
    for (int r = 0; r < 8; ++r)
      sS[(mb + r) * T + n] = (half_t)sc[r];
  }

  // ---- row softmax (one lane per query row; wave-lockstep, DScnt-ordered)
  if (lane < 16) {
    half_t* row = sS + lane * T;
    float m = -1e30f;
    for (int j = 0; j < T; ++j) { float v = (float)row[j]; m = v > m ? v : m; }
    float sum = 0.f;
    for (int j = 0; j < T; ++j) {
      float e = __expf((float)row[j] - m);
      sum += e;
      row[j] = (half_t)e;
    }
    float inv = 1.f / sum;
    for (int j = 0; j < T; ++j) row[j] = (half_t)((float)row[j] * inv);
  }

#if HAVE_TDM
  __builtin_amdgcn_s_wait_tensorcnt(0);     // V tile resident in LDS
#endif

  // ---- O = P @ V over key chunks of 32 (two 16-channel output tiles)
  v8f acc0 = {}, acc1 = {};
  for (int kc = 0; kc < T; kc += 32) {
    const half_t* prow = sS + (lane & 15) * T + kc + ((lane >> 4) << 4);
    u16h pa;
    pa.s.lo = *(const v8h*)(prow);          // ds_load_b128
    pa.s.hi = *(const v8h*)(prow + 8);
    int kb = kc + ((lane >> 4) << 4);
#if HAVE_TDM
    const half_t* v0 = vL + (size_t)(lane & 15) * T + kb;
    const half_t* v1 = vL + (size_t)(16 + (lane & 15)) * T + kb;
#else
    const half_t* v0 = vbase + (size_t)(lane & 15) * T + kb;
    const half_t* v1 = vbase + (size_t)(16 + (lane & 15)) * T + kb;
#endif
    u16h ub0, ub1;
    ub0.s.lo = *(const v8h*)(v0); ub0.s.hi = *(const v8h*)(v0 + 8);
    ub1.s.lo = *(const v8h*)(v1); ub1.s.hi = *(const v8h*)(v1 + 8);
    acc0 = wmma_f16(pa.v, ub0.v, acc0);
    acc1 = wmma_f16(pa.v, ub1.v, acc1);
  }

  // ---- store with lepe add and inverse roll
  int wpr = HWDIM / wsp;
  int mb  = (lane >> 4) << 3;
  int c0  = lane & 15;
  for (int r = 0; r < 8; ++r) {
    int tok = qb + mb + r;
    int hr = (win / wpr) * hsp + tok / wsp;
    int wr = (win % wpr) * wsp + tok % wsp;
    int h  = (hr + (hsp >> 1)) & 255;
    int x  = (wr + (wsp >> 1)) & 255;
    int p  = (h << 8) | x;
    out[(size_t)c0 * NPIX + p]        = acc0[r] + lepe[(size_t)c0 * NPIX + p];
    out[(size_t)(16 + c0) * NPIX + p] = acc1[r] + lepe[(size_t)(16 + c0) * NPIX + p];
  }
}

// ---------------------------------------------------------------------------
__global__ void __launch_bounds__(256)
k_add(const float* __restrict__ a, const float* __restrict__ b,
      float* __restrict__ out, int n) {
  int i = blockIdx.x * 256 + threadIdx.x;
  if (i < n) out[i] = a[i] + b[i];
}

__global__ void __launch_bounds__(256)
k_add_gelu(const float* __restrict__ a, const float* __restrict__ b,
           float* __restrict__ out, int n) {
  int i = blockIdx.x * 256 + threadIdx.x;
  if (i < n) out[i] = gelu_f(a[i] + b[i]);
}

__global__ void __launch_bounds__(256)
k_gate_mix(float* __restrict__ h, const float* __restrict__ z,
           const float* __restrict__ hh, int n) {
  int i = blockIdx.x * 256 + threadIdx.x;
  if (i < n) { float zz = z[i]; h[i] = (1.f - zz) * h[i] + zz * hh[i]; }
}

// ---------------------------------------------------------------------------
// Input index map: x first, then params in JAX pytree (sorted-key) order.
// attnN (base 1+10*br): k5_b,k5_w,lepe_b,lepe_w,q5_b,q5_w,qk_b,qk_w,v_b,v_w
// dwc (41): c1_b,c1_w,c2_b,c2_w,c3_b,c3_w,c4a_b,c4a_w,c4b_b,c4b_w
// 51 fc1_b 52 fc1_w 53 fc2_b 54 fc2_w
// gate (55): f_b,f_w,r_b,r_w,z_b,z_w
// 61 n1_b 62 n1_w 63 n2_b 64 n2_w 65 proj1_b 66 proj1_w 67 proj_b 68 proj_w
extern "C" void kernel_launch(void* const* d_in, const int* in_sizes, int n_in,
                              void* d_out, int out_size, void* d_ws, size_t ws_size,
                              hipStream_t stream) {
  (void)in_sizes; (void)n_in; (void)out_size; (void)ws_size;
  auto F = [&](int i) { return (const float*)d_in[i]; };

  char* wsb = (char*)d_ws;
  size_t cur = 0;
  auto alloc = [&](size_t bytes) -> char* {
    char* p = wsb + cur;
    cur = (cur + bytes + 255) & ~(size_t)255;
    return p;
  };

  // f16 weight pool
  half_t* w_proj1 = (half_t*)alloc(128 * 64 * 2);
  half_t* w_qk[4]; half_t* w_v[4];
  for (int br = 0; br < 4; ++br) {
    w_qk[br] = (half_t*)alloc(64 * 32 * 2);
    w_v[br]  = (half_t*)alloc(32 * 32 * 2);
  }
  half_t* w_f    = (half_t*)alloc(32 * 64 * 2);
  half_t* w_r    = (half_t*)alloc(32 * 64 * 2);
  half_t* w_z    = (half_t*)alloc(32 * 64 * 2);
  half_t* w_proj = (half_t*)alloc(64 * 128 * 2);
  half_t* w_fc1  = (half_t*)alloc(128 * 64 * 2);
  half_t* w_fc2  = (half_t*)alloc(64 * 128 * 2);

  // staging / activations
  half_t* xn16    = (half_t*)alloc((size_t)64 * NPIX * 2);
  half_t* s16     = (half_t*)alloc((size_t)32 * NPIX * 2);
  half_t* qw16    = (half_t*)alloc((size_t)32 * NPIX * 2);
  half_t* kw16    = (half_t*)alloc((size_t)32 * NPIX * 2);
  half_t* vw16    = (half_t*)alloc((size_t)32 * NPIX * 2);
  half_t* pack128 = (half_t*)alloc((size_t)128 * NPIX * 2);
  float* y1    = (float*)alloc((size_t)128 * NPIX * 4);
  float* qkbuf = (float*)alloc((size_t)64 * NPIX * 4);
  float* vbuf  = (float*)alloc((size_t)32 * NPIX * 4);
  float* lepeb = (float*)alloc((size_t)32 * NPIX * 4);
  float* q5b   = (float*)alloc((size_t)32 * NPIX * 4);
  float* k5b   = (float*)alloc((size_t)32 * NPIX * 4);
  float* xbr[4];
  for (int br = 0; br < 4; ++br) xbr[br] = (float*)alloc((size_t)32 * NPIX * 4);
  float* rbuf  = (float*)alloc((size_t)32 * NPIX * 4);
  float* zbuf  = (float*)alloc((size_t)32 * NPIX * 4);
  float* hhbuf = (float*)alloc((size_t)32 * NPIX * 4);
  float* ybuf  = (float*)alloc((size_t)64 * NPIX * 4);
  float* xmid  = (float*)alloc((size_t)64 * NPIX * 4);
  float* hbuf  = (float*)alloc((size_t)128 * NPIX * 4);
  float* hbuf2 = (float*)alloc((size_t)128 * NPIX * 4);

  auto cvt = [&](int idx, half_t* dst, int n) {
    k_f32_to_f16<<<(n + 255) / 256, 256, 0, stream>>>(F(idx), dst, n);
  };
  cvt(66, w_proj1, 128 * 64);
  for (int br = 0; br < 4; ++br) {
    cvt(1 + 10 * br + 7, w_qk[br], 64 * 32);
    cvt(1 + 10 * br + 9, w_v[br], 32 * 32);
  }
  cvt(56, w_f, 32 * 64);
  cvt(58, w_r, 32 * 64);
  cvt(60, w_z, 32 * 64);
  cvt(68, w_proj, 64 * 128);
  cvt(52, w_fc1, 128 * 64);
  cvt(54, w_fc2, 64 * 128);

  const float* x_in = F(0);
  const int EW32  = (32 * NPIX + 255) / 256;
  const int EW64  = (64 * NPIX + 255) / 256;
  const int EW128 = (128 * NPIX + 255) / 256;

  auto gemm = [&](const half_t* W, const half_t* X, const float* bias,
                  float* Y, int M, int K, int act) {
    int tiles = (M >> 4) * (NPIX >> 4);   // always a multiple of 8
    k_gemm16<<<tiles / 8, 256, 0, stream>>>(W, X, bias, Y, M, NPIX, K, act);
  };

  // LN1 + proj1 (dim -> 2*dim)
  k_ln_pack<<<NPIX / 256, 256, 0, stream>>>(x_in, F(62), F(61), xn16);
  gemm(w_proj1, xn16, F(65), y1, 128, 64, 0);

  // four LePE attention branches
  const int HSP[4] = {32, 8, 32, 8};
  const int WSP[4] = {32, 32, 8, 32};
  const float scale = 0.17677669529663687f;  // 32^-0.5
  for (int br = 0; br < 4; ++br) {
    int base = 1 + 10 * br;
    k_pack_pm<<<EW32, 256, 0, stream>>>(y1 + (size_t)br * 32 * NPIX, nullptr,
                                        s16, 32, 32, 0);
    gemm(w_qk[br], s16, F(base + 6), qkbuf, 64, 32, 0);
    gemm(w_v[br],  s16, F(base + 8), vbuf,  32, 32, 0);
    k_dwconv<<<EW32, 256, 0, stream>>>(vbuf, nullptr, F(base + 3), F(base + 2),
                                       lepeb, 32, 5, 5, 0);
    k_dwconv<<<EW32, 256, 0, stream>>>(qkbuf, nullptr, F(base + 5), F(base + 4),
                                       q5b, 32, 5, 5, 0);
    k_dwconv<<<EW32, 256, 0, stream>>>(qkbuf + (size_t)32 * NPIX, nullptr,
                                       F(base + 1), F(base + 0), k5b, 32, 5, 5, 0);
    k_attn_pack<<<NPIX / 256, 256, 0, stream>>>(q5b, k5b, lepeb, vbuf,
                                                qw16, kw16, vw16,
                                                HSP[br], WSP[br], scale);
    k_attn<<<NPIX / 16, 32, 0, stream>>>(qw16, kw16, vw16, lepeb, xbr[br],
                                         HSP[br], WSP[br]);
  }

  // gated merges: x2 = gate(x1, x2); x4 = gate(x3, x4)
  auto gate = [&](float* xa, float* xh) {
    k_pack_pm<<<EW32, 256, 0, stream>>>(xa, nullptr, pack128, 32, 64, 0);
    k_pack_pm<<<EW32, 256, 0, stream>>>(xh, nullptr, pack128, 32, 64, 32);
    gemm(w_r, pack128, F(57), rbuf, 32, 64, 1);
    gemm(w_z, pack128, F(59), zbuf, 32, 64, 1);
    k_pack_pm<<<EW32, 256, 0, stream>>>(xh, rbuf, pack128, 32, 64, 32);  // h*r
    gemm(w_f, pack128, F(55), hhbuf, 32, 64, 2);
    k_gate_mix<<<EW32, 256, 0, stream>>>(xh, zbuf, hhbuf, 32 * NPIX);
  };
  gate(xbr[0], xbr[1]);
  gate(xbr[2], xbr[3]);

  // proj(concat) + residual
  for (int br = 0; br < 4; ++br)
    k_pack_pm<<<EW32, 256, 0, stream>>>(xbr[br], nullptr, pack128, 32, 128, 32 * br);
  gemm(w_proj, pack128, F(67), ybuf, 64, 128, 0);
  k_add<<<EW64, 256, 0, stream>>>(x_in, ybuf, xmid, 64 * NPIX);

  // LN2 + fc1 + dconv_mlp + gelu + fc2 + residual
  k_ln_pack<<<NPIX / 256, 256, 0, stream>>>(xmid, F(64), F(63), xn16);
  gemm(w_fc1, xn16, F(51), hbuf, 128, 64, 0);
  k_dwconv<<<EW32, 256, 0, stream>>>(hbuf, nullptr, F(42), F(41),
                                     hbuf2, 32, 3, 3, 1);
  k_dwconv<<<EW32, 256, 0, stream>>>(hbuf + (size_t)32 * NPIX, hbuf2, F(44), F(43),
                                     hbuf2 + (size_t)32 * NPIX, 32, 5, 3, 1);
  k_dwconv<<<EW32, 256, 0, stream>>>(hbuf + (size_t)64 * NPIX,
                                     hbuf2 + (size_t)32 * NPIX, F(46), F(45),
                                     hbuf2 + (size_t)64 * NPIX, 32, 3, 5, 1);
  k_dwconv<<<EW32, 256, 0, stream>>>(hbuf + (size_t)96 * NPIX,
                                     hbuf2 + (size_t)64 * NPIX, F(48), F(47),
                                     q5b, 32, 3, 3, 0);
  k_dwconv<<<EW32, 256, 0, stream>>>(q5b, nullptr, F(50), F(49),
                                     hbuf2 + (size_t)96 * NPIX, 32, 3, 3, 1);
  k_add_gelu<<<EW128, 256, 0, stream>>>(hbuf2, hbuf, hbuf, 128 * NPIX);
  k_pack_pm<<<EW128, 256, 0, stream>>>(hbuf, nullptr, pack128, 128, 128, 0);
  gemm(w_fc2, pack128, F(53), ybuf, 64, 128, 0);
  k_add<<<EW64, 256, 0, stream>>>(xmid, ybuf, (float*)d_out, 64 * NPIX);
}
